// EncodeProcessDecode_8126078124039
// MI455X (gfx1250) — compile-verified, matrix operand order checked by reading, not run on hardware
//
#include <hip/hip_runtime.h>

// ---------------------------------------------------------------------------
// EncodeProcessDecode GNN on MI455X (gfx1250, wave32, WMMA).
// bf16 activations/weights, f32 accumulation via v_wmma_f32_16x16x32_bf16.
// Each MLP (3 layers + LN) fused in one kernel: 128-row tile staged into LDS
// via async global->LDS DMA (gather concat included), three WMMA GEMM layers
// per-wave, wave-cooperative LayerNorm + residual + fused float-atomic
// segment-sum scatter.
// ---------------------------------------------------------------------------

typedef __attribute__((ext_vector_type(16))) __bf16 v16bf;
typedef __attribute__((ext_vector_type(8)))  __bf16 v8bf;
typedef __attribute__((ext_vector_type(8)))  float  v8f;
typedef __attribute__((ext_vector_type(4)))  int    v4i;

#define N_NODES 50000
#define N_EDGES 400000
#define LATENT  128
#define OUTF    8
#define STEPS   4
#define LN_EPS  1e-5f
#define MT      128   // rows per workgroup (8 waves x 16 rows)

enum { MODE_ENC = 0, MODE_EDGE = 1, MODE_NODE = 2, MODE_DEC = 3 };
enum { OUT_ENC_NODE = 0, OUT_ENC_EDGE = 1, OUT_RESID_EDGE = 2,
       OUT_RESID_NODE = 3, OUT_DEC = 4 };

// ---- CDNA5 async global->LDS copy (ASYNCcnt path), guarded ----------------
#if defined(__has_builtin)
#if __has_builtin(__builtin_amdgcn_global_load_async_to_lds_b128) && \
    __has_builtin(__builtin_amdgcn_s_wait_asynccnt)
#define USE_ASYNC_LDS 1
#endif
#endif
#ifndef USE_ASYNC_LDS
#define USE_ASYNC_LDS 0
#endif

__device__ __forceinline__ void async_copy16(const void* gsrc, void* ldst) {
#if USE_ASYNC_LDS
  // Builtin takes (AS1 int4* src, AS3 dst, imm offset, imm cpol).
  __builtin_amdgcn_global_load_async_to_lds_b128(
      (__attribute__((address_space(1))) v4i*)
          (__attribute__((address_space(1))) void*)(void*)gsrc,
      (__attribute__((address_space(3))) v4i*)
          (__attribute__((address_space(3))) void*)ldst,
      0, 0);
#else
  *(v8bf*)ldst = *(const v8bf*)gsrc;
#endif
}

__device__ __forceinline__ void async_join() {
#if USE_ASYNC_LDS
  __builtin_amdgcn_s_wait_asynccnt(0);
#endif
}

// ---- weight prep: f32 W[in][out] -> bf16 Wt[outpad][kpad] (zero padded) ----
__global__ void wprep_kernel(const float* __restrict__ W, __bf16* __restrict__ Wt,
                             int in_dim, int out_dim, int kpad, int outpad) {
  int i = blockIdx.x * 256 + threadIdx.x;
  int total = outpad * kpad;
  if (i >= total) return;
  int o = i / kpad, k = i - o * kpad;
  float v = (o < out_dim && k < in_dim) ? W[(long)k * out_dim + o] : 0.f;
  Wt[i] = (__bf16)v;
}

__global__ void fill0_kernel(float* __restrict__ p, long n) {
  long i = (long)blockIdx.x * 256 + threadIdx.x;
  if (i < n) p[i] = 0.f;
}

// ---- WMMA fragment load: bf16 row chunk [k..k+7] and [k+16..k+23] ----------
__device__ __forceinline__ v16bf load_frag(const __bf16* p) {
  v8bf lo = *(const v8bf*)p;
  v8bf hi = *(const v8bf*)(p + 16);
  v16bf r;
#pragma unroll
  for (int i = 0; i < 8; ++i) { r[i] = lo[i]; r[i + 8] = hi[i]; }
  return r;
}

__device__ __forceinline__ v8bf zero8bf() {
  v8bf z;
#pragma unroll
  for (int j = 0; j < 8; ++j) z[j] = (__bf16)0.f;
  return z;
}

// ---- per-wave GEMM: 16 rows x (NT*16) cols, K1 multiple of 32 --------------
template <int NT>
__device__ __forceinline__ void gemm_rows16(const __bf16* __restrict__ A, int K1,
                                            const __bf16* __restrict__ Wt,
                                            int w, int lane_lo, int hi, v8f* acc) {
  v8f zf;
#pragma unroll
  for (int j = 0; j < 8; ++j) zf[j] = 0.f;
#pragma unroll
  for (int t = 0; t < NT; ++t) acc[t] = zf;

  const __bf16* arow = A + ((w * 16 + lane_lo) * K1 + hi * 8);
  for (int k0 = 0; k0 < K1; k0 += 32) {
    v16bf a = load_frag(arow + k0);
#pragma unroll
    for (int t = 0; t < NT; ++t) {
      const __bf16* brow = Wt + ((long)(t * 16 + lane_lo) * K1 + k0 + hi * 8);
      v16bf b = load_frag(brow);
      acc[t] = __builtin_amdgcn_wmma_f32_16x16x32_bf16(
          false, a, false, b, (short)0, acc[t], false, false);
    }
  }
}

// ---- bias + relu + bf16 store of hidden layer into LDS ---------------------
__device__ __forceinline__ void store_hidden(__bf16* __restrict__ Hb, const v8f* acc,
                                             int w, int lane_lo, int hi,
                                             const float* __restrict__ bias) {
#pragma unroll
  for (int t = 0; t < 8; ++t) {
    int col = t * 16 + lane_lo;
    float bv = bias[col];
#pragma unroll
    for (int r = 0; r < 8; ++r) {
      int row = w * 16 + r + hi * 8;
      float v = acc[t][r] + bv;
      v = fmaxf(v, 0.f);
      Hb[row * LATENT + col] = (__bf16)v;
    }
  }
}

// ---------------------------------------------------------------------------
// Fused 3-layer MLP (+LN, +residual, +scatter) kernel.
// ---------------------------------------------------------------------------
__global__ void __launch_bounds__(256)
mlp_fused_kernel(int mode, int outmode, long nrows, int K1, int in_raw, int abytes,
                 const float* __restrict__ xin,
                 const __bf16* __restrict__ nodes_bf_in,
                 const __bf16* __restrict__ edges_bf_in,
                 const float* __restrict__ agg_in,
                 const int* __restrict__ senders,
                 const int* __restrict__ receivers,
                 const __bf16* __restrict__ Wt1, const float* __restrict__ b1,
                 const __bf16* __restrict__ Wt2, const float* __restrict__ b2,
                 const __bf16* __restrict__ Wt3, const float* __restrict__ b3,
                 const float* __restrict__ ln_s, const float* __restrict__ ln_b,
                 float* __restrict__ nodes_f32,
                 __bf16* __restrict__ nodes_bf_out,
                 __bf16* __restrict__ edges_bf_out,
                 float* __restrict__ agg_out,
                 float* __restrict__ dec_out) {
  extern __shared__ unsigned char smem[];
  __bf16* A  = (__bf16*)smem;            // MT x K1 bf16 input tile
  float*  Hf = (float*)smem;             // MT x 128 f32 (overlays A after L1)
  __bf16* Hb = (__bf16*)(smem + abytes); // MT x 128 bf16 hidden

  const int tid = threadIdx.x;
  const int w = tid >> 5;
  const int lane = tid & 31;
  const int lane_lo = lane & 15;
  const int hi = lane >> 4;
  const long row0 = (long)blockIdx.x * MT;

  // pull weight streams toward the caches (global_prefetch_b8)
  __builtin_prefetch(Wt1, 0, 0);
  __builtin_prefetch(Wt2, 0, 0);
  __builtin_prefetch(Wt3, 0, 0);

  // -------- stage A tile into LDS (bf16, zero-padded) --------
  if (mode == MODE_ENC) {
    for (int i = tid; i < MT * 32; i += 256) {
      int lr = i >> 5, c = i & 31;
      long g = row0 + lr;
      float v = (g < nrows && c < in_raw) ? xin[g * in_raw + c] : 0.f;
      A[lr * 32 + c] = (__bf16)v;
    }
  } else if (mode == MODE_EDGE) {          // concat [send | recv | edge], K1=384
    for (int i = tid; i < MT * 48; i += 256) {
      int lr = i / 48, c = i - lr * 48;
      long g = row0 + lr;
      __bf16* dst = A + lr * 384 + c * 8;
      if (g < nrows) {
        const __bf16* src;
        if (c < 16)
          src = nodes_bf_in + (long)senders[g] * LATENT + c * 8;
        else if (c < 32)
          src = nodes_bf_in + (long)receivers[g] * LATENT + (c - 16) * 8;
        else
          src = edges_bf_in + g * LATENT + (c - 32) * 8;
        async_copy16(src, dst);            // global_load_async_to_lds_b128
      } else {
        *(v8bf*)dst = zero8bf();
      }
    }
  } else if (mode == MODE_NODE) {          // concat [nodes | agg], K1=256
    for (int i = tid; i < MT * 32; i += 256) {
      int lr = i >> 5, c = i & 31;
      long g = row0 + lr;
      __bf16* dst = A + lr * 256 + c * 8;
      if (g < nrows && c < 16) {
        async_copy16(nodes_bf_in + g * LATENT + c * 8, dst);
      } else if (g < nrows) {              // agg half: f32 -> bf16 convert
        const float* ap = agg_in + g * LATENT + (c - 16) * 8;
        v8bf val;
#pragma unroll
        for (int j = 0; j < 8; ++j) val[j] = (__bf16)ap[j];
        *(v8bf*)dst = val;
      } else {
        *(v8bf*)dst = zero8bf();
      }
    }
  } else {                                  // MODE_DEC, K1=128
    for (int i = tid; i < MT * 16; i += 256) {
      int lr = i >> 4, c = i & 15;
      long g = row0 + lr;
      __bf16* dst = A + lr * LATENT + c * 8;
      if (g < nrows) async_copy16(nodes_bf_in + g * LATENT + c * 8, dst);
      else           *(v8bf*)dst = zero8bf();
    }
  }
  async_join();       // s_wait_asynccnt 0 (wave's DMA into LDS complete)
  __syncthreads();

  v8f acc[8];

  // -------- layer 1 --------
  gemm_rows16<8>(A, K1, Wt1, w, lane_lo, hi, acc);
  __syncthreads();   // all A reads complete (Hf overlays the A region later)
  store_hidden(Hb, acc, w, lane_lo, hi, b1);
  __syncthreads();

  // -------- layer 2 (in-place on Hb; each wave owns its 16 rows) --------
  gemm_rows16<8>(Hb, LATENT, Wt2, w, lane_lo, hi, acc);
  __syncthreads();
  store_hidden(Hb, acc, w, lane_lo, hi, b2);
  __syncthreads();

  if (outmode == OUT_DEC) {
    // -------- decoder layer 3: 128 -> 8 (Wt3 padded to [16][128]) --------
    v8f acc3[1];
    gemm_rows16<1>(Hb, LATENT, Wt3, w, lane_lo, hi, acc3);
    int col = lane_lo;
    if (col < OUTF) {
      float bv = b3[col];
#pragma unroll
      for (int r = 0; r < 8; ++r) {
        long g = row0 + w * 16 + r + hi * 8;
        if (g < nrows) dec_out[g * OUTF + col] = acc3[0][r] + bv;
      }
    }
    return;
  }

  // -------- layer 3 -> f32 tile in Hf (overlays A) --------
  gemm_rows16<8>(Hb, LATENT, Wt3, w, lane_lo, hi, acc);
#pragma unroll
  for (int t = 0; t < 8; ++t) {
    int col = t * 16 + lane_lo;
    float bv = b3[col];
#pragma unroll
    for (int r = 0; r < 8; ++r) {
      int row = w * 16 + r + hi * 8;
      Hf[row * LATENT + col] = acc[t][r] + bv;
    }
  }
  __syncthreads();

  // -------- LayerNorm + residual/scatter, wave-cooperative ----------------
  // Each wave handles its 16 rows; per row all 32 lanes reduce sum/sum^2
  // via wave32 shuffles, then each lane emits 4 columns.
  for (int rr = 0; rr < 16; ++rr) {
    int row = w * 16 + rr;
    long g = row0 + row;
    if (g >= nrows) continue;              // uniform within wave
    const float* hr = Hf + row * LATENT;
    float v[4];
    float s = 0.f, ss = 0.f;
#pragma unroll
    for (int j = 0; j < 4; ++j) {
      v[j] = hr[lane + 32 * j];
      s += v[j];
      ss += v[j] * v[j];
    }
#pragma unroll
    for (int m = 16; m >= 1; m >>= 1) {
      s  += __shfl_xor(s, m, 32);
      ss += __shfl_xor(ss, m, 32);
    }
    float mu = s * (1.f / LATENT);
    float var = ss * (1.f / LATENT) - mu * mu;
    float rstd = rsqrtf(var + LN_EPS);

    if (outmode == OUT_ENC_NODE) {
#pragma unroll
      for (int j = 0; j < 4; ++j) {
        int c = lane + 32 * j;
        float h = (v[j] - mu) * rstd * ln_s[c] + ln_b[c];
        nodes_f32[g * LATENT + c] = h;
        nodes_bf_out[g * LATENT + c] = (__bf16)h;
      }
    } else if (outmode == OUT_ENC_EDGE) {
#pragma unroll
      for (int j = 0; j < 4; ++j) {
        int c = lane + 32 * j;
        float h = (v[j] - mu) * rstd * ln_s[c] + ln_b[c];
        edges_bf_out[g * LATENT + c] = (__bf16)h;
      }
    } else if (outmode == OUT_RESID_EDGE) {
      long rcv = (long)receivers[g] * LATENT;
#pragma unroll
      for (int j = 0; j < 4; ++j) {
        int c = lane + 32 * j;
        float ne = (v[j] - mu) * rstd * ln_s[c] + ln_b[c];
        atomicAdd(&agg_out[rcv + c], ne);                   // segment_sum
        float e = (float)edges_bf_out[g * LATENT + c] + ne; // edge residual
        edges_bf_out[g * LATENT + c] = (__bf16)e;
      }
    } else { // OUT_RESID_NODE
#pragma unroll
      for (int j = 0; j < 4; ++j) {
        int c = lane + 32 * j;
        float nn = (v[j] - mu) * rstd * ln_s[c] + ln_b[c];
        float nf = nodes_f32[g * LATENT + c] + nn;          // node residual
        nodes_f32[g * LATENT + c] = nf;
        nodes_bf_out[g * LATENT + c] = (__bf16)nf;
      }
    }
  }
}

// ---------------------------------------------------------------------------
// Host orchestration.
// d_in order: node_features, edge_features, senders, receivers,
//   enc_node{W0,W1,W2,b0,b1,b2,ln_s,ln_b} (4..11),
//   enc_edge{...} (12..19),
//   dec{W0,W1,W2,b0,b1,b2} (20..25, ln=None dropped),
//   step_edge[s] at 26+8s, step_node[s] at 58+8s.
// ---------------------------------------------------------------------------
extern "C" void kernel_launch(void* const* d_in, const int* in_sizes, int n_in,
                              void* d_out, int out_size, void* d_ws, size_t ws_size,
                              hipStream_t stream) {
  (void)in_sizes; (void)n_in; (void)out_size; (void)ws_size;

  char* ws = (char*)d_ws;
  size_t off = 0;
  auto alloc = [&](size_t bytes) -> char* {
    char* p = ws + off;
    off += (bytes + 255) & ~(size_t)255;
    return p;
  };

  float*  nodes_f32 = (float*)alloc((size_t)N_NODES * LATENT * 4);
  __bf16* nodes_bf  = (__bf16*)alloc((size_t)N_NODES * LATENT * 2);
  __bf16* edges_bf  = (__bf16*)alloc((size_t)N_EDGES * LATENT * 2);
  float*  agg       = (float*)alloc((size_t)N_NODES * LATENT * 4);

  auto F = [&](int i) { return (const float*)d_in[i]; };
  const float* node_feat = F(0);
  const float* edge_feat = F(1);
  const int* senders   = (const int*)d_in[2];
  const int* receivers = (const int*)d_in[3];

  auto prep = [&](const float* W, int in, int out, int kpad, int outpad) -> __bf16* {
    __bf16* dst = (__bf16*)alloc((size_t)kpad * outpad * 2);
    int total = kpad * outpad;
    wprep_kernel<<<(total + 255) / 256, 256, 0, stream>>>(W, dst, in, out, kpad, outpad);
    return dst;
  };

  __bf16 *Wt_encn[3], *Wt_ence[3], *Wt_dec[3], *Wt_se[STEPS][3], *Wt_sn[STEPS][3];
  Wt_encn[0] = prep(F(4), 16, 128, 32, 128);
  Wt_encn[1] = prep(F(5), 128, 128, 128, 128);
  Wt_encn[2] = prep(F(6), 128, 128, 128, 128);
  Wt_ence[0] = prep(F(12), 4, 128, 32, 128);
  Wt_ence[1] = prep(F(13), 128, 128, 128, 128);
  Wt_ence[2] = prep(F(14), 128, 128, 128, 128);
  Wt_dec[0]  = prep(F(20), 128, 128, 128, 128);
  Wt_dec[1]  = prep(F(21), 128, 128, 128, 128);
  Wt_dec[2]  = prep(F(22), 128, 8, 128, 16);
  for (int s = 0; s < STEPS; ++s) {
    int be = 26 + 8 * s, bn = 58 + 8 * s;
    Wt_se[s][0] = prep(F(be + 0), 384, 128, 384, 128);
    Wt_se[s][1] = prep(F(be + 1), 128, 128, 128, 128);
    Wt_se[s][2] = prep(F(be + 2), 128, 128, 128, 128);
    Wt_sn[s][0] = prep(F(bn + 0), 256, 128, 256, 128);
    Wt_sn[s][1] = prep(F(bn + 1), 128, 128, 128, 128);
    Wt_sn[s][2] = prep(F(bn + 2), 128, 128, 128, 128);
  }

  auto launch_mlp = [&](int mode, int outmode, long nrows, int K1, int in_raw,
                        const float* xin, const float* agg_in,
                        const __bf16* Wt1, const float* b1,
                        const __bf16* Wt2, const float* b2,
                        const __bf16* Wt3, const float* b3,
                        const float* lns, const float* lnb, float* dec_o) {
    int abytes = MT * K1 * 2;
    if (abytes < MT * LATENT * 4) abytes = MT * LATENT * 4;  // room for f32 Hf overlay
    size_t smem = (size_t)abytes + (size_t)MT * LATENT * 2;  // + bf16 Hb
    int blocks = (int)((nrows + MT - 1) / MT);
    mlp_fused_kernel<<<blocks, 256, smem, stream>>>(
        mode, outmode, nrows, K1, in_raw, abytes,
        xin, nodes_bf, edges_bf, agg_in, senders, receivers,
        Wt1, b1, Wt2, b2, Wt3, b3, lns, lnb,
        nodes_f32, nodes_bf, edges_bf, agg, dec_o);
  };

  // Encoders
  launch_mlp(MODE_ENC, OUT_ENC_NODE, N_NODES, 32, 16, node_feat, nullptr,
             Wt_encn[0], F(7), Wt_encn[1], F(8), Wt_encn[2], F(9), F(10), F(11), nullptr);
  launch_mlp(MODE_ENC, OUT_ENC_EDGE, N_EDGES, 32, 4, edge_feat, nullptr,
             Wt_ence[0], F(15), Wt_ence[1], F(16), Wt_ence[2], F(17), F(18), F(19), nullptr);

  // Processor steps
  for (int s = 0; s < STEPS; ++s) {
    long aggN = (long)N_NODES * LATENT;
    fill0_kernel<<<(int)((aggN + 255) / 256), 256, 0, stream>>>(agg, aggN);
    int be = 26 + 8 * s, bn = 58 + 8 * s;
    launch_mlp(MODE_EDGE, OUT_RESID_EDGE, N_EDGES, 384, 0, nullptr, nullptr,
               Wt_se[s][0], F(be + 3), Wt_se[s][1], F(be + 4), Wt_se[s][2], F(be + 5),
               F(be + 6), F(be + 7), nullptr);
    launch_mlp(MODE_NODE, OUT_RESID_NODE, N_NODES, 256, 0, nullptr, agg,
               Wt_sn[s][0], F(bn + 3), Wt_sn[s][1], F(bn + 4), Wt_sn[s][2], F(bn + 5),
               F(bn + 6), F(bn + 7), nullptr);
  }

  // Decoder
  launch_mlp(MODE_DEC, OUT_DEC, N_NODES, 128, 0, nullptr, nullptr,
             Wt_dec[0], F(23), Wt_dec[1], F(24), Wt_dec[2], F(25),
             nullptr, nullptr, (float*)d_out);
}
